// DynFKhot_33389075759176
// MI455X (gfx1250) — matrix operand order repaired and checked by reading, not compiled, not run on hardware
//
#include <hip/hip_runtime.h>

// Problem sizes (match reference)
#define BSZ   32768
#define DDIM  512
#define HDIM  512
#define QDIM  1024
#define DQDIM 1536

typedef __bf16 bf16_t;
typedef bf16_t        v16bf __attribute__((ext_vector_type(16)));
typedef float         v8f   __attribute__((ext_vector_type(8)));
typedef float         v4f   __attribute__((ext_vector_type(4)));
typedef unsigned int  v4u   __attribute__((ext_vector_type(4)));

union Frag { v4u q[2]; v16bf v; };

__device__ __forceinline__ unsigned short f2bf(float f) {
  unsigned u = __float_as_uint(f);
  u += 0x7FFFu + ((u >> 16) & 1u);          // round-to-nearest-even
  return (unsigned short)(u >> 16);
}
__device__ __forceinline__ float bf2f(unsigned short h) {
  return __uint_as_float(((unsigned)h) << 16);
}

// ---------------------------------------------------------------------------
// fp32 -> bf16 elementwise convert (8 elems/thread, b128 in / b128 out)
// ---------------------------------------------------------------------------
__global__ __launch_bounds__(256) void cvt_f32_bf16(const float* __restrict__ src,
                                                    unsigned short* __restrict__ dst) {
  const int i = (blockIdx.x * 256 + threadIdx.x) * 8;
  v4f a = *(const v4f*)(src + i);
  v4f b = *(const v4f*)(src + i + 4);
  v4u o;
  o[0] = (unsigned)f2bf(a[0]) | ((unsigned)f2bf(a[1]) << 16);
  o[1] = (unsigned)f2bf(a[2]) | ((unsigned)f2bf(a[3]) << 16);
  o[2] = (unsigned)f2bf(b[0]) | ((unsigned)f2bf(b[1]) << 16);
  o[3] = (unsigned)f2bf(b[2]) | ((unsigned)f2bf(b[3]) << 16);
  *(v4u*)(dst + i) = o;
}

// ---------------------------------------------------------------------------
// fp32 [K,N] row-major  ->  bf16 [N,K] row-major (transposed weights)
// grid = (N/32, K/32), block = 256 (32x8 logical)
// ---------------------------------------------------------------------------
__global__ __launch_bounds__(256) void transpose_cvt(const float* __restrict__ W,
                                                     unsigned short* __restrict__ WT,
                                                     int K, int N) {
  __shared__ float tile[32][33];
  const int tx = threadIdx.x & 31;
  const int ty = threadIdx.x >> 5;            // 0..7
  const int k0 = blockIdx.y * 32;
  const int n0 = blockIdx.x * 32;
#pragma unroll
  for (int i = 0; i < 4; ++i)
    tile[ty + i * 8][tx] = W[(size_t)(k0 + ty + i * 8) * N + n0 + tx];
  __syncthreads();
#pragma unroll
  for (int i = 0; i < 4; ++i)
    WT[(size_t)(n0 + ty + i * 8) * K + k0 + tx] = f2bf(tile[tx][ty + i * 8]);
}

// ---------------------------------------------------------------------------
// bf16 WMMA GEMM:  out[M,N] = act( A[M,Ktot] @ BT[N,Ktot]^T + bias )
// A is split across two sources (A0 for K<K0, A1 after) to fuse the concat.
// Block: 256 threads (8 waves), tile 128x128, K-tile 64 (2 x K=32 substeps).
// Wave w: rows (w>>1)*32..+31, cols (w&1)*64..+63  -> 2x4 WMMA accumulators.
// ---------------------------------------------------------------------------
__global__ __launch_bounds__(256)
void gemm_bf16_wmma(const unsigned short* __restrict__ A0, int lda0, int K0,
                    const unsigned short* __restrict__ A1, int lda1, int K1,
                    const unsigned short* __restrict__ BT, int ldb,
                    const float* __restrict__ bias,
                    float* __restrict__ outF, int ldoF,
                    unsigned short* __restrict__ outH, int ldoH,
                    int relu) {
  const int tid  = threadIdx.x;
  const int wave = tid >> 5;
  const int lane = tid & 31;
  const int wm = wave >> 1;          // 0..3
  const int wn = wave & 1;           // 0..1
  const int lm = lane & 15;
  const int hi = lane >> 4;          // lane-group (K half select)

  const int mbase = blockIdx.y * 128;
  const int nbase = blockIdx.x * 128;

  // LDS staging [128 x 64] bf16, +8 bf16 row pad (stride 72 u16 = 144 B)
  __shared__ unsigned short As[128 * 72];
  __shared__ unsigned short Bs[128 * 72];

  v8f acc[2][4];
#pragma unroll
  for (int mt = 0; mt < 2; ++mt)
#pragma unroll
    for (int nt = 0; nt < 4; ++nt)
      acc[mt][nt] = (v8f){0.f, 0.f, 0.f, 0.f, 0.f, 0.f, 0.f, 0.f};

  const int Ktot = K0 + K1;
  const int r  = tid >> 1;           // 0..127 staging row
  const int hf = tid & 1;            // 0/1 -> K sub-offset 32

  for (int kb = 0; kb < Ktot; kb += 64) {
    const unsigned short* aSrc;
    int alda, acol;
    if (kb < K0) { aSrc = A0; alda = lda0; acol = kb; }
    else         { aSrc = A1; alda = lda1; acol = kb - K0; }

    // Stage A block [128 x 64] (32 bf16 per thread, 4x b128 each side)
    {
      const v4u* gp = (const v4u*)(aSrc + (size_t)(mbase + r) * alda + acol + hf * 32);
      v4u a0 = gp[0], a1 = gp[1], a2 = gp[2], a3 = gp[3];
      v4u* lp = (v4u*)&As[r * 72 + hf * 32];
      lp[0] = a0; lp[1] = a1; lp[2] = a2; lp[3] = a3;
      // Stage B block [128 cols x 64 K] from transposed weights
      const v4u* gb = (const v4u*)(BT + (size_t)(nbase + r) * ldb + kb + hf * 32);
      v4u b0 = gb[0], b1 = gb[1], b2 = gb[2], b3 = gb[3];
      v4u* lq = (v4u*)&Bs[r * 72 + hf * 32];
      lq[0] = b0; lq[1] = b1; lq[2] = b2; lq[3] = b3;
      // Prefetch next K block into near caches (WGP scope, high locality)
      const int kn = kb + 64;
      if (kn < Ktot) {
        const unsigned short* pa; int plda, pcol;
        if (kn < K0) { pa = A0; plda = lda0; pcol = kn; }
        else         { pa = A1; plda = lda1; pcol = kn - K0; }
        __builtin_prefetch(pa + (size_t)(mbase + r) * plda + pcol + hf * 32, 0, 3);
        __builtin_prefetch(BT + (size_t)(nbase + r) * ldb + kn + hf * 32, 0, 3);
      }
    }
    __syncthreads();

    // Two K=32 substeps; fragments per ISA VGPR layouts (16-bit A 16x32 / B 32x16)
#pragma unroll
    for (int ks = 0; ks < 64; ks += 32) {
      Frag aF[2], bF[4];
#pragma unroll
      for (int mt = 0; mt < 2; ++mt) {
        const int row = wm * 32 + mt * 16 + lm;
        aF[mt].q[0] = *(const v4u*)&As[row * 72 + ks + hi * 8];       // K 8hi..+7
        aF[mt].q[1] = *(const v4u*)&As[row * 72 + ks + 16 + hi * 8];  // K 16+8hi..+7
      }
#pragma unroll
      for (int nt = 0; nt < 4; ++nt) {
        const int nn = wn * 64 + nt * 16 + lm;
        bF[nt].q[0] = *(const v4u*)&Bs[nn * 72 + ks + hi * 8];        // K 8hi..+7
        bF[nt].q[1] = *(const v4u*)&Bs[nn * 72 + ks + 16 + hi * 8];   // K 16+8hi..+7
      }
#pragma unroll
      for (int mt = 0; mt < 2; ++mt)
#pragma unroll
        for (int nt = 0; nt < 4; ++nt)
          acc[mt][nt] = __builtin_amdgcn_wmma_f32_16x16x32_bf16(
              false, aF[mt].v, false, bF[nt].v, (short)0, acc[mt][nt], false, false);
    }
    __syncthreads();
  }

  // Epilogue: bias + optional ReLU, optional f32 and/or bf16 outputs
#pragma unroll
  for (int mt = 0; mt < 2; ++mt)
#pragma unroll
    for (int nt = 0; nt < 4; ++nt) {
      const int col  = nbase + wn * 64 + nt * 16 + lm;
      const float bc = bias ? bias[col] : 0.0f;
      const int rowb = mbase + wm * 32 + mt * 16 + hi * 8;
#pragma unroll
      for (int e = 0; e < 8; ++e) {
        float v = acc[mt][nt][e] + bc;
        if (relu) v = fmaxf(v, 0.0f);
        const size_t row = (size_t)(rowb + e);
        if (outF) outF[row * ldoF + col] = v;
        if (outH) outH[row * ldoH + col] = f2bf(v);
      }
    }
}

// ---------------------------------------------------------------------------
// k predictor layer 3: k[b] = clip(sigmoid(h2[b]·w3 + b3)*Q * sigmoid(ks)*2, 1, Q)
// one wave per row; 512-elem dot; shuffle reduction
// ---------------------------------------------------------------------------
__global__ __launch_bounds__(256)
void kpred_kernel(const unsigned short* __restrict__ h2, const float* __restrict__ w3,
                  const float* __restrict__ b3, const float* __restrict__ kscale,
                  float* __restrict__ kvals, float* __restrict__ kout) {
  const int wave = threadIdx.x >> 5, lane = threadIdx.x & 31;
  const int b = blockIdx.x * 8 + wave;
  const v4u* rp = (const v4u*)(h2 + (size_t)b * HDIM + lane * 16);
  v4u h0 = rp[0], h1 = rp[1];                       // 16 bf16
  const v4f* wp = (const v4f*)(w3 + lane * 16);
  v4f w0 = wp[0], w1 = wp[1], w2 = wp[2], w3v = wp[3];
  float s = 0.f;
#pragma unroll
  for (int i = 0; i < 4; ++i) {
    s += bf2f((unsigned short)(h0[i] & 0xFFFFu)) * ((i < 2) ? w0[2 * i] : w1[2 * i - 4]);
    s += bf2f((unsigned short)(h0[i] >> 16)) * ((i < 2) ? w0[2 * i + 1] : w1[2 * i - 3]);
  }
#pragma unroll
  for (int i = 0; i < 4; ++i) {
    s += bf2f((unsigned short)(h1[i] & 0xFFFFu)) * ((i < 2) ? w2[2 * i] : w3v[2 * i - 4]);
    s += bf2f((unsigned short)(h1[i] >> 16)) * ((i < 2) ? w2[2 * i + 1] : w3v[2 * i - 3]);
  }
#pragma unroll
  for (int off = 16; off >= 1; off >>= 1) s += __shfl_xor(s, off, 32);
  if (lane == 0) {
    const float z  = s + b3[0];
    const float kr = (float)QDIM / (1.0f + __expf(-z));
    const float sc = 1.0f / (1.0f + __expf(-kscale[0]));
    float kk = kr * sc * 2.0f;
    kk = fminf(fmaxf(kk, 1.0f), (float)QDIM);
    kvals[b] = kk;
    kout[b]  = kk;
  }
}

// ---------------------------------------------------------------------------
// Exact per-row top-k mask: 4-pass radix select over order-preserving u32 keys,
// stable tie-break by index (matches stable argsort rank < k, count = ceil(k)).
// One 256-thread block per row; 4 contiguous elements per thread.
// ---------------------------------------------------------------------------
__global__ __launch_bounds__(256)
void topk_kernel(const float* __restrict__ logits, const float* __restrict__ kvals,
                 float* __restrict__ out) {
  const int b = blockIdx.x;
  const int t = threadIdx.x;
  __shared__ unsigned hist[256];
  __shared__ unsigned scanbuf[256];
  __shared__ unsigned sh_prefix, sh_rem;

  v4f xv = *(const v4f*)(logits + (size_t)b * QDIM + t * 4);
  unsigned k4[4];
#pragma unroll
  for (int e = 0; e < 4; ++e) {
    unsigned u = __float_as_uint(xv[e]);
    k4[e] = (u & 0x80000000u) ? ~u : (u | 0x80000000u);  // ascending-order key
  }
  if (t == 0) {
    int cnt = (int)ceilf(kvals[b]);
    cnt = cnt < 1 ? 1 : (cnt > QDIM ? QDIM : cnt);
    sh_rem = (unsigned)cnt;
    sh_prefix = 0u;
  }
  __syncthreads();

  for (int p = 0; p < 4; ++p) {
    const int shift = 24 - 8 * p;
    hist[t] = 0u;
    __syncthreads();
    const unsigned prefix = sh_prefix;
#pragma unroll
    for (int e = 0; e < 4; ++e) {
      const unsigned key = k4[e];
      const bool cand = (p == 0) || ((key >> (shift + 8)) == (prefix >> (shift + 8)));
      if (cand) atomicAdd(&hist[(key >> shift) & 255u], 1u);
    }
    __syncthreads();
    if (t == 0) {
      unsigned remL = sh_rem, cum = 0u;
      int sel = 0;
      for (int bin = 255; bin >= 0; --bin) {
        const unsigned h = hist[bin];
        if (cum + h >= remL) { sel = bin; remL -= cum; break; }
        cum += h;
      }
      sh_prefix = prefix | ((unsigned)sel << shift);
      sh_rem = remL;
    }
    __syncthreads();
  }
  const unsigned T = sh_prefix;
  const unsigned rem = sh_rem;

  unsigned eqc = 0;
  bool gt[4], eq[4];
#pragma unroll
  for (int e = 0; e < 4; ++e) {
    gt[e] = k4[e] > T;
    eq[e] = (k4[e] == T);
    eqc += eq[e] ? 1u : 0u;
  }
  scanbuf[t] = eqc;
  __syncthreads();
  for (int off = 1; off < 256; off <<= 1) {   // inclusive scan (index order)
    const unsigned v = scanbuf[t];
    const unsigned add = (t >= off) ? scanbuf[t - off] : 0u;
    __syncthreads();
    scanbuf[t] = v + add;
    __syncthreads();
  }
  unsigned run = scanbuf[t] - eqc;            // exclusive prefix of equals
  v4f o;
#pragma unroll
  for (int e = 0; e < 4; ++e) {
    const bool take = gt[e] || (eq[e] && run < rem);
    if (eq[e]) run++;
    o[e] = take ? 1.0f : 0.0f;
  }
  *(v4f*)(out + (size_t)b * QDIM + t * 4) = o;
}

// ---------------------------------------------------------------------------
extern "C" void kernel_launch(void* const* d_in, const int* in_sizes, int n_in,
                              void* d_out, int out_size, void* d_ws, size_t ws_size,
                              hipStream_t stream) {
  const float* x      = (const float*)d_in[0];
  const float* enc_w1 = (const float*)d_in[1];
  const float* enc_b1 = (const float*)d_in[2];
  const float* enc_w2 = (const float*)d_in[3];
  const float* enc_b2 = (const float*)d_in[4];
  const float* kp_w1  = (const float*)d_in[5];
  const float* kp_b1  = (const float*)d_in[6];
  const float* kp_w2  = (const float*)d_in[7];
  const float* kp_b2  = (const float*)d_in[8];
  const float* kp_w3  = (const float*)d_in[9];
  const float* kp_b3  = (const float*)d_in[10];
  const float* kscale = (const float*)d_in[11];
  float* outp = (float*)d_out;

  unsigned char* ws = (unsigned char*)d_ws;
  unsigned short* x_bf  = (unsigned short*)(ws + 0);           //  32 MB [B,512]
  unsigned short* hid1  = (unsigned short*)(ws + 33554432);    //  32 MB [B,512]
  unsigned short* logbf = (unsigned short*)(ws + 67108864);    //  64 MB [B,1024]
  float*          logF  = (float*)        (ws + 134217728);    // 128 MB [B,1024]
  unsigned short* kph1  = (unsigned short*)(ws + 268435456);   //  32 MB [B,512]
  unsigned short* kph2  = (unsigned short*)(ws + 301989888);   //  32 MB [B,512]
  unsigned short* w1T   = (unsigned short*)(ws + 335544320);   // [512,512]
  unsigned short* w2T   = (unsigned short*)(ws + 336068608);   // [1024,512]
  unsigned short* kp1T  = (unsigned short*)(ws + 337117184);   // [512,1536]
  unsigned short* kp2T  = (unsigned short*)(ws + 338690048);   // [512,512]
  float*          kvals = (float*)        (ws + 339214336);    // [B]

  // 1) Precision prep: x -> bf16, weights -> transposed bf16
  cvt_f32_bf16<<<dim3((BSZ * DDIM) / 2048), dim3(256), 0, stream>>>(x, x_bf);
  transpose_cvt<<<dim3(HDIM / 32, DDIM / 32), dim3(256), 0, stream>>>(enc_w1, w1T, DDIM, HDIM);
  transpose_cvt<<<dim3(QDIM / 32, HDIM / 32), dim3(256), 0, stream>>>(enc_w2, w2T, HDIM, QDIM);
  transpose_cvt<<<dim3(HDIM / 32, DQDIM / 32), dim3(256), 0, stream>>>(kp_w1, kp1T, DQDIM, HDIM);
  transpose_cvt<<<dim3(HDIM / 32, HDIM / 32), dim3(256), 0, stream>>>(kp_w2, kp2T, HDIM, HDIM);

  // 2) Encoder: hid1 = relu(x@w1+b1) ; logits = hid1@w2+b2 (fp32 + bf16 copies)
  gemm_bf16_wmma<<<dim3(HDIM / 128, BSZ / 128), dim3(256), 0, stream>>>(
      x_bf, DDIM, DDIM, nullptr, 0, 0, w1T, DDIM, enc_b1,
      nullptr, 0, hid1, HDIM, 1);
  gemm_bf16_wmma<<<dim3(QDIM / 128, BSZ / 128), dim3(256), 0, stream>>>(
      hid1, HDIM, HDIM, nullptr, 0, 0, w2T, HDIM, enc_b2,
      logF, QDIM, logbf, QDIM, 0);

  // 3) k predictor: fused concat via split-A GEMM, then two more layers
  gemm_bf16_wmma<<<dim3(HDIM / 128, BSZ / 128), dim3(256), 0, stream>>>(
      x_bf, DDIM, DDIM, logbf, QDIM, QDIM, kp1T, DQDIM, kp_b1,
      nullptr, 0, kph1, HDIM, 1);
  gemm_bf16_wmma<<<dim3(HDIM / 128, BSZ / 128), dim3(256), 0, stream>>>(
      kph1, HDIM, HDIM, nullptr, 0, 0, kp2T, HDIM, kp_b2,
      nullptr, 0, kph2, HDIM, 1);
  kpred_kernel<<<dim3(BSZ / 8), dim3(256), 0, stream>>>(
      kph2, kp_w3, kp_b3, kscale, kvals, outp + (size_t)BSZ * QDIM);

  // 4) Exact top-k mask from fp32 logits
  topk_kernel<<<dim3(BSZ), dim3(256), 0, stream>>>(logF, kvals, outp);
}
